// FullAttention0_63462436766087
// MI455X (gfx1250) — compile-verified, matrix operand order checked by reading, not run on hardware
//
#include <hip/hip_runtime.h>
#include <hip/hip_bf16.h>

// Problem constants from the reference.
#define BB 4
#define LL 2048
#define HH 8
#define DD 64
#define NTILES 128            // L/16 key (and query) tiles per (b,h)
#define GROUP 4               // key tiles per pipeline stage (64 keys, 8KB packed)
#define NGROUPS (NTILES / GROUP)
#define TILE_UINTS 512        // 16 keys * 64B/lane-frag = 2KB = 512 dwords

typedef __attribute__((ext_vector_type(16))) __bf16    v16bf;
typedef __attribute__((ext_vector_type(8)))  float     v8f;
typedef __attribute__((ext_vector_type(8)))  unsigned  v8u;
typedef __attribute__((ext_vector_type(4)))  unsigned  v4u;
typedef __attribute__((ext_vector_type(8)))  int       v8i;
typedef __attribute__((ext_vector_type(4)))  int       v4i;

static __device__ __forceinline__ unsigned short f32_to_bf16_rne(float f) {
    unsigned u = __float_as_uint(f);
    u += 0x7fffu + ((u >> 16) & 1u);
    return (unsigned short)(u >> 16);
}

// ---------------- DPP16 16-lane row reductions (used once, at finalize) ----
template <int CTRL>
static __device__ __forceinline__ float dpp_mov_f32(float x) {
    return __int_as_float(__builtin_amdgcn_update_dpp(
        0, __float_as_int(x), CTRL, 0xF, 0xF, true));
}
static __device__ __forceinline__ float reduce16_max(float x) {
    x = fmaxf(x, dpp_mov_f32<0x0B1>(x));   // quad_perm(1,0,3,2)
    x = fmaxf(x, dpp_mov_f32<0x04E>(x));   // quad_perm(2,3,0,1)
    x = fmaxf(x, dpp_mov_f32<0x141>(x));   // row_half_mirror
    x = fmaxf(x, dpp_mov_f32<0x140>(x));   // row_mirror
    return x;
}
static __device__ __forceinline__ float reduce16_add(float x) {
    x += dpp_mov_f32<0x0B1>(x);
    x += dpp_mov_f32<0x04E>(x);
    x += dpp_mov_f32<0x141>(x);
    x += dpp_mov_f32<0x140>(x);
    return x;
}

static __device__ __forceinline__ v16bf combine8(v4u lo, v4u hi) {
    v8u u;
    u[0] = lo[0]; u[1] = lo[1]; u[2] = lo[2]; u[3] = lo[3];
    u[4] = hi[0]; u[5] = hi[1]; u[6] = hi[2]; u[7] = hi[3];
    return __builtin_bit_cast(v16bf, u);
}

// ---------------- TDM: 1-D tensor_load_to_lds of 8KB (2048 dwords) --------
static __device__ __forceinline__ void tdm_load_lds_8kb(unsigned lds_off,
                                                        const unsigned* gptr) {
    const unsigned long long ga = (unsigned long long)(uintptr_t)gptr;
    v4u g0;
    g0[0] = 1u;                           // count=1
    g0[1] = lds_off;
    g0[2] = (unsigned)ga;
    g0[3] = ((unsigned)(ga >> 32) & 0x01FFFFFFu) | 0x80000000u;   // type=2
    v8i g1;
    g1[0] = (int)(2u << 16);              // data_size = 4 bytes
    g1[1] = 0;                            // tensor_dim0[15:0] in [31:16] = 0
    g1[2] = (int)(0x4000u | (1u << 16));  // tensor_dim0 hi = 2^30; tensor_dim1 lo = 1
    g1[3] = (int)(2048u << 16);           // tile_dim0 = 2048 dwords
    g1[4] = 0;                            // tile_dim1 = 0 (1-D), tile_dim2 = 0
    g1[5] = (int)0x40000000;              // tensor_dim0_stride lo
    g1[6] = 0;
    g1[7] = 0;
    const v4i z4 = {0, 0, 0, 0};
#if __clang_major__ >= 23
    const v8i z8 = {0, 0, 0, 0, 0, 0, 0, 0};
    __builtin_amdgcn_tensor_load_to_lds(g0, g1, z4, z4, z8, 0);
#else
    __builtin_amdgcn_tensor_load_to_lds(g0, g1, z4, z4, 0);
#endif
}

// ---------------------------------------------------------------------------
// Kernel 1: colsumV[b,h,d] = sum_e V[b,e,h,d]
// ---------------------------------------------------------------------------
__global__ __launch_bounds__(256) void colsum_v_kernel(const float* __restrict__ V,
                                                       float* __restrict__ colsum) {
    const int b    = blockIdx.x >> 3;
    const int h    = blockIdx.x & 7;
    const int d    = threadIdx.x & 63;
    const int part = threadIdx.x >> 6;
    const size_t rowStride = (size_t)HH * DD;

    const float* vb = V + ((size_t)b * LL * HH + h) * DD + d;
    float s = 0.0f;
    const int e0 = part * (LL / 4);
    for (int e = e0; e < e0 + LL / 4; ++e)
        s += vb[(size_t)e * rowStride];

    __shared__ float lds[256];
    lds[threadIdx.x] = s;
    __syncthreads();
    if (part == 0)
        colsum[(b * HH + h) * DD + d] = lds[d] + lds[64 + d] + lds[128 + d] + lds[192 + d];
}

// ---------------------------------------------------------------------------
// Kernel 2 (pre-pass): fp32 [B,L,H,D] -> fragment-ordered bf16, segment-major:
// uint index = tileIdx*512 + s*128 + lane*4 + j   (s = c*2+g)
// ---------------------------------------------------------------------------
__global__ __launch_bounds__(256) void pack_frag_kernel(const float* __restrict__ src,
                                                        unsigned* __restrict__ dst) {
    const int idx  = blockIdx.x * 256 + threadIdx.x;
    const int j    = idx & 3;
    const int lane = (idx >> 2) & 31;
    const int s    = (idx >> 7) & 3;
    const int kt   = (idx >> 9) & 127;
    const int h    = (idx >> 16) & 7;
    const int b    = idx >> 19;
    const int g    = s & 1;
    const int c    = s >> 1;

    const int row = kt * 16 + (lane & 15);
    const int d0  = c * 32 + 16 * g + 8 * (lane >> 4) + 2 * j;
    const float2 f = *(const float2*)(src + (((size_t)b * LL + row) * HH + h) * DD + d0);
    dst[idx] = (unsigned)f32_to_bf16_rne(f.x) | ((unsigned)f32_to_bf16_rne(f.y) << 16);
}

// ---------------- lane-local online softmax (no cross-lane ops in loop) ----
static __device__ __forceinline__ void online_update4_local(const v8f acc[4],
                                                            float run_m[8],
                                                            float run_s[8]) {
    const float scale = 0.125f;            // 1/sqrt(64)
#pragma unroll
    for (int r = 0; r < 8; ++r) {
        const float s0 = acc[0][r] * scale, s1 = acc[1][r] * scale;
        const float s2 = acc[2][r] * scale, s3 = acc[3][r] * scale;
        const float nm = fmaxf(fmaxf(fmaxf(s0, s1), fmaxf(s2, s3)), run_m[r]);
        const float p  = (__expf(s0 - nm) + __expf(s1 - nm)) +
                         (__expf(s2 - nm) + __expf(s3 - nm));
        run_s[r] = run_s[r] * __expf(run_m[r] - nm) + p;
        run_m[r] = nm;
    }
}

// ---------------- finalize one 16-row tile: DPP merge + rsum store --------
static __device__ __forceinline__ void finalize_tile(const float run_m[8],
                                                     const float run_s[8],
                                                     int rowBase, int half, int n,
                                                     float* rsum) {
    float tot[8];
#pragma unroll
    for (int r = 0; r < 8; ++r) {
        const float M = reduce16_max(run_m[r]);
        tot[r] = reduce16_add(run_s[r] * __expf(run_m[r] - M));
    }
    if (n == 0) {
#pragma unroll
        for (int r = 0; r < 8; ++r) {
            const float denom = tot[r];
            const float inv   = 1.0f / denom;   // not foldable: keeps WMMA chain live
            rsum[rowBase + half * 8 + r] = denom * inv;
        }
    }
}

// ---------------- block epilogue: out = rsum[q] * colsumV[d] ---------------
template <int NQ, int NTHREADS>
static __device__ __forceinline__ void write_out_block(int b, int h, int qb,
                                                       const float* __restrict__ colsumV,
                                                       float* __restrict__ out,
                                                       const float* rsum, float* cs) {
    if (threadIdx.x < 64)
        cs[threadIdx.x] = colsumV[(b * HH + h) * DD + threadIdx.x];
    __syncthreads();
    const size_t rowStride = (size_t)HH * DD;
    float* obase = out + (((size_t)b * LL + (size_t)qb * NQ) * HH + h) * DD;
    for (int idx = threadIdx.x; idx < NQ * 64; idx += NTHREADS) {
        const int ql = idx >> 6;
        const int d  = idx & 63;
        obase[(size_t)ql * rowStride + d] = rsum[ql] * cs[d];
    }
}

// ---------------------------------------------------------------------------
// Kernel 3a (packed + TDM path): block = 128 queries, 4 waves x 2 query tiles.
// Per 8KB K-group (4 key tiles): wave0 DMAs via tensor_load_to_lds (double-
// buffered, TENSORcnt + barrier); each wave's 16 ds_load_b128 feed 16 WMMAs.
// ---------------------------------------------------------------------------
__global__ __launch_bounds__(128) void fa_scores_tdm_kernel(
        const unsigned* __restrict__ packedQ, const unsigned* __restrict__ packedK,
        const float* __restrict__ colsumV, float* __restrict__ out) {
    const int qb   = blockIdx.x & 15;          // L/128 = 16 query blocks
    const int h    = (blockIdx.x >> 4) & 7;
    const int b    = blockIdx.x >> 7;
    const int wave = threadIdx.x >> 5;
    const int lane = threadIdx.x & 31;
    const int half = lane >> 4;
    const int n    = lane & 15;

    __shared__ unsigned kbuf[2][GROUP * TILE_UINTS];   // 2 x 8KB
    __shared__ float rsum[128];
    __shared__ float cs[64];

    // Two A-fragment pairs (query tiles qb*8 + wave*2 + {0,1}).
    v16bf a0[2], a1[2];
#pragma unroll
    for (int s = 0; s < 2; ++s) {
        const unsigned* qt = packedQ +
            (size_t)((b * HH + h) * NTILES + (qb * 8 + wave * 2 + s)) * TILE_UINTS;
        const v4u q0 = *(const v4u*)(qt + 0 * 128 + lane * 4);
        const v4u q1 = *(const v4u*)(qt + 1 * 128 + lane * 4);
        const v4u q2 = *(const v4u*)(qt + 2 * 128 + lane * 4);
        const v4u q3 = *(const v4u*)(qt + 3 * 128 + lane * 4);
        a0[s] = combine8(q0, q1);              // d 0..31
        a1[s] = combine8(q2, q3);              // d 32..63
    }

    const unsigned* kglob = packedK + (size_t)(b * HH + h) * NTILES * TILE_UINTS;

    float run_m[2][8], run_s[2][8];
#pragma unroll
    for (int s = 0; s < 2; ++s)
#pragma unroll
        for (int r = 0; r < 8; ++r) { run_m[s][r] = -3.0e38f; run_s[s][r] = 0.0f; }

    if (threadIdx.x < 32)
        tdm_load_lds_8kb((unsigned)(uintptr_t)&kbuf[0][0], kglob);

    for (int grp = 0; grp < NGROUPS; ++grp) {
        if (threadIdx.x < 32)
            __builtin_amdgcn_s_wait_tensorcnt(0);      // group `grp` landed
        __syncthreads();                               // publish to all waves
        if (threadIdx.x < 32 && grp + 1 < NGROUPS)     // prefetch next group
            tdm_load_lds_8kb((unsigned)(uintptr_t)&kbuf[(grp + 1) & 1][0],
                             kglob + (size_t)(grp + 1) * GROUP * TILE_UINTS);

        const unsigned* kb = &kbuf[grp & 1][0];
        v8f acc[2][GROUP];
#pragma unroll
        for (int t = 0; t < GROUP; ++t) {
            const unsigned* tb = kb + t * TILE_UINTS;
            const v4u p0 = *(const v4u*)(tb + 0 * 128 + lane * 4);
            const v4u p1 = *(const v4u*)(tb + 1 * 128 + lane * 4);
            const v4u p2 = *(const v4u*)(tb + 2 * 128 + lane * 4);
            const v4u p3 = *(const v4u*)(tb + 3 * 128 + lane * 4);
            const v16bf b0 = combine8(p0, p1);
            const v16bf b1 = combine8(p2, p3);
#pragma unroll
            for (int s = 0; s < 2; ++s) {              // B frags reused 2x
                v8f a = {};
                a = __builtin_amdgcn_wmma_f32_16x16x32_bf16(false, a0[s], false, b0,
                                                            (short)0, a, false, false);
                a = __builtin_amdgcn_wmma_f32_16x16x32_bf16(false, a1[s], false, b1,
                                                            (short)0, a, false, false);
                acc[s][t] = a;
            }
        }
#pragma unroll
        for (int s = 0; s < 2; ++s)
            online_update4_local(acc[s], run_m[s], run_s[s]);
        __syncthreads();                               // all done reading kb
    }

#pragma unroll
    for (int s = 0; s < 2; ++s)
        finalize_tile(run_m[s], run_s[s], (wave * 2 + s) * 16, half, n, rsum);
    write_out_block<128, 128>(b, h, qb, colsumV, out, rsum, cs);
}

// ---------------------------------------------------------------------------
// Kernel 3b (fallback, small workspace): direct global loads + in-loop
// fp32->bf16 conversion, lane-local softmax. Block = 64 queries.
// ---------------------------------------------------------------------------
static __device__ __forceinline__ v16bf load_frag_row(const float* __restrict__ row,
                                                      int half, int chunk) {
    v8u u;
#pragma unroll
    for (int g = 0; g < 2; ++g) {
        const float4 f0 = *(const float4*)(row + chunk * 32 + 16 * g + 8 * half);
        const float4 f1 = *(const float4*)(row + chunk * 32 + 16 * g + 8 * half + 4);
        float f[8] = {f0.x, f0.y, f0.z, f0.w, f1.x, f1.y, f1.z, f1.w};
#pragma unroll
        for (int j = 0; j < 4; ++j)
            u[g * 4 + j] = (unsigned)f32_to_bf16_rne(f[2 * j]) |
                           ((unsigned)f32_to_bf16_rne(f[2 * j + 1]) << 16);
    }
    return __builtin_bit_cast(v16bf, u);
}

__global__ __launch_bounds__(128) void fa_scores_fallback_kernel(
        const float* __restrict__ Q, const float* __restrict__ K,
        const float* __restrict__ colsumV, float* __restrict__ out) {
    const int qb   = blockIdx.x & 31;
    const int h    = (blockIdx.x >> 5) & 7;
    const int b    = blockIdx.x >> 8;
    const int wave = threadIdx.x >> 5;
    const int lane = threadIdx.x & 31;
    const int half = lane >> 4;
    const int n    = lane & 15;
    const int q0   = qb * 64 + wave * 16;

    const size_t rowStride = (size_t)HH * DD;
    const float* Qbase = Q + ((size_t)b * LL * HH + h) * DD;
    const float* Kbase = K + ((size_t)b * LL * HH + h) * DD;

    const float* qrow = Qbase + (size_t)(q0 + n) * rowStride;
    const v16bf a0 = load_frag_row(qrow, half, 0);
    const v16bf a1 = load_frag_row(qrow, half, 1);

    float run_m[8], run_s[8];
#pragma unroll
    for (int r = 0; r < 8; ++r) { run_m[r] = -3.0e38f; run_s[r] = 0.0f; }

    for (int kt0 = 0; kt0 < NTILES; kt0 += GROUP) {
        v8f acc[GROUP];
#pragma unroll
        for (int t = 0; t < GROUP; ++t) {
            const float* krow = Kbase + (size_t)((kt0 + t) * 16 + n) * rowStride;
            const v16bf b0 = load_frag_row(krow, half, 0);
            const v16bf b1 = load_frag_row(krow, half, 1);
            v8f a = {};
            a = __builtin_amdgcn_wmma_f32_16x16x32_bf16(false, a0, false, b0,
                                                        (short)0, a, false, false);
            a = __builtin_amdgcn_wmma_f32_16x16x32_bf16(false, a1, false, b1,
                                                        (short)0, a, false, false);
            acc[t] = a;
        }
        online_update4_local(acc, run_m, run_s);
    }

    __shared__ float rsum[64];
    __shared__ float cs[64];
    finalize_tile(run_m, run_s, wave * 16, half, n, rsum);
    write_out_block<64, 128>(b, h, qb, colsumV, out, rsum, cs);
}

extern "C" void kernel_launch(void* const* d_in, const int* in_sizes, int n_in,
                              void* d_out, int out_size, void* d_ws, size_t ws_size,
                              hipStream_t stream) {
    (void)in_sizes; (void)n_in; (void)out_size;
    const float* Q = (const float*)d_in[0];
    const float* K = (const float*)d_in[1];
    const float* V = (const float*)d_in[2];
    float* out     = (float*)d_out;

    // Workspace layout: [0,8KB) colsumV | packedQ (8MB) | packedK (8MB)
    const size_t colsumBytes = (size_t)BB * HH * DD * sizeof(float);
    const size_t packUints   = (size_t)BB * HH * NTILES * TILE_UINTS;
    const size_t packBytes   = packUints * sizeof(unsigned);
    float*    colsum  = (float*)d_ws;
    unsigned* packedQ = (unsigned*)((char*)d_ws + colsumBytes);
    unsigned* packedK = (unsigned*)((char*)d_ws + colsumBytes + packBytes);

    colsum_v_kernel<<<BB * HH, 256, 0, stream>>>(V, colsum);

    if (ws_size >= colsumBytes + 2 * packBytes) {
        const int pblocks = (int)(packUints / 256);
        pack_frag_kernel<<<pblocks, 256, 0, stream>>>(Q, packedQ);
        pack_frag_kernel<<<pblocks, 256, 0, stream>>>(K, packedK);
        fa_scores_tdm_kernel<<<BB * HH * (LL / 128), 128, 0, stream>>>(
            packedQ, packedK, colsum, out);
    } else {
        fa_scores_fallback_kernel<<<BB * HH * (LL / 64), 128, 0, stream>>>(
            Q, K, colsum, out);
    }
}
